// MMDIT_78726750536137
// MI455X (gfx1250) — matrix operand order repaired and checked by reading, not compiled, not run on hardware
//
#include <hip/hip_runtime.h>
#include <hip/hip_bf16.h>

#define B_ 4
#define D_ 1024
#define H_ 16
#define HD_ 64
#define N1_ 1024
#define N2_ 64
#define NT_ 1088
#define FRAMES_ 16

typedef _Float16 v16h __attribute__((ext_vector_type(16)));
typedef _Float16 h8   __attribute__((ext_vector_type(8)));
typedef _Float16 h4   __attribute__((ext_vector_type(4)));
typedef float    v8f  __attribute__((ext_vector_type(8)));

static __device__ __forceinline__ v16h cat16(h8 lo, h8 hi) {
  union { v16h v; h8 h[2]; } u; u.h[0] = lo; u.h[1] = hi; return u.v;
}
static __device__ __forceinline__ v8f wmma16(v16h a, v16h b, v8f c) {
  // D = A(16x32 f16) * B(32x16 f16) + C(16x16 f32)
  return __builtin_amdgcn_wmma_f32_16x16x32_f16(false, a, false, b, (short)0, c, false, false);
}
static __device__ __forceinline__ int frame_of(int t) {
  return t < N1_ ? (t >> 6) : ((t - N1_) >> 2);
}
static __device__ __forceinline__ float gelu_t(float x) {
  float x3 = x * x * x;
  return 0.5f * x * (1.0f + tanhf(0.7978845608f * (x + 0.044715f * x3)));
}

// ---------------------------------------------------------------------------
// Generic WMMA GEMM: C[M,N] = A[M,K](f16) * W[N,K](f32->f16)^T + bias
// EPI 0: fp32 out.  EPI 1: gelu -> f16 out.
// M % 128 == 0, N % 64 == 0, K % 32 == 0 (guaranteed by the problem sizes).
// ---------------------------------------------------------------------------
template <int EPI>
__global__ __launch_bounds__(256) void gemm_wmma(
    const _Float16* __restrict__ A, const float* __restrict__ W,
    const float* __restrict__ bias, float* __restrict__ Cf,
    _Float16* __restrict__ Ch, int M, int N, int K) {
  __shared__ _Float16 As[128 * 40];
  __shared__ _Float16 Bs[64 * 40];
  const int tid = threadIdx.x;
  const int m0 = blockIdx.y * 128, n0 = blockIdx.x * 64;
  const int w = tid >> 5, lane = tid & 31;
  const int wm = w >> 1, wn = w & 1;
  const int l15 = lane & 15, g = lane >> 4;

  v8f zero = {0.f, 0.f, 0.f, 0.f, 0.f, 0.f, 0.f, 0.f};
  v8f acc[2][2];
  acc[0][0] = zero; acc[0][1] = zero; acc[1][0] = zero; acc[1][1] = zero;

  for (int k0 = 0; k0 < K; k0 += 32) {
    // stage A tile 128x32 f16
#pragma unroll
    for (int c = 0; c < 2; ++c) {
      int qd = c * 256 + tid;
      int row = qd >> 2, kc = qd & 3;
      const _Float16* src = A + (size_t)(m0 + row) * K + k0 + kc * 8;
      if (k0 + 32 < K) __builtin_prefetch(src + 32, 0, 1);
      *(h8*)&As[row * 40 + kc * 8] = *(const h8*)src;
    }
    // stage W tile 64x32, fp32 -> f16
#pragma unroll
    for (int c = 0; c < 2; ++c) {
      int qd = c * 256 + tid;
      int row = qd >> 3, kc = qd & 7;
      const float* srcw = W + (size_t)(n0 + row) * K + k0 + kc * 4;
      if (k0 + 32 < K) __builtin_prefetch(srcw + 32, 0, 1);
      float4 wv = *(const float4*)srcw;
      h4 hv = {(_Float16)wv.x, (_Float16)wv.y, (_Float16)wv.z, (_Float16)wv.w};
      *(h4*)&Bs[row * 40 + kc * 4] = hv;
    }
    __syncthreads();

    v16h af[2], bf[2];
#pragma unroll
    for (int mt = 0; mt < 2; ++mt) {
      int row = wm * 32 + mt * 16 + l15;
      af[mt] = cat16(*(const h8*)&As[row * 40 + g * 8],
                     *(const h8*)&As[row * 40 + 16 + g * 8]);
    }
#pragma unroll
    for (int nt = 0; nt < 2; ++nt) {
      int col = wn * 32 + nt * 16 + l15;
      bf[nt] = cat16(*(const h8*)&Bs[col * 40 + g * 8],
                     *(const h8*)&Bs[col * 40 + 16 + g * 8]);
    }
#pragma unroll
    for (int mt = 0; mt < 2; ++mt)
#pragma unroll
      for (int nt = 0; nt < 2; ++nt)
        acc[mt][nt] = wmma16(af[mt], bf[nt], acc[mt][nt]);
    __syncthreads();
  }

#pragma unroll
  for (int mt = 0; mt < 2; ++mt)
#pragma unroll
    for (int nt = 0; nt < 2; ++nt)
#pragma unroll
      for (int r = 0; r < 8; ++r) {
        int m = m0 + wm * 32 + mt * 16 + r + g * 8;
        int n = n0 + wn * 32 + nt * 16 + l15;
        float v = acc[mt][nt][r] + bias[n];
        if (EPI == 0) Cf[(size_t)m * N + n] = v;
        else          Ch[(size_t)m * N + n] = (_Float16)gelu_t(v);
      }
}

// ---------------------------------------------------------------------------
// Flash attention with WMMA: one wave per (b*H+h, 16-row Q tile).
// q,k: (b,h,token,d) f16.  vT: (b,h,d,token) f16.  Block-causal frame mask.
// ---------------------------------------------------------------------------
__global__ __launch_bounds__(32) void flash_attn(
    const _Float16* __restrict__ q, const _Float16* __restrict__ k,
    const _Float16* __restrict__ vT, _Float16* __restrict__ o1,
    _Float16* __restrict__ o2) {
  __shared__ _Float16 P[16 * 40];
  const int bh = blockIdx.x;
  const int qt = blockIdx.y;
  const int lane = threadIdx.x & 31;
  const int l15 = lane & 15, g = lane >> 4;
  const int q0 = qt * 16;
  const size_t qk0 = (size_t)bh * NT_ * HD_;

  v16h Aq[2];
#pragma unroll
  for (int s = 0; s < 2; ++s) {
    const _Float16* p = q + qk0 + (size_t)(q0 + l15) * HD_ + s * 32 + g * 8;
    Aq[s] = cat16(*(const h8*)p, *(const h8*)(p + 16));
  }
  int fq[8];
#pragma unroll
  for (int r = 0; r < 8; ++r) fq[r] = frame_of(q0 + r + g * 8);
  const int maxfq = frame_of(q0 + 15);

  v8f zero = {0.f, 0.f, 0.f, 0.f, 0.f, 0.f, 0.f, 0.f};
  v8f Oa[4];
  float rmax[8], rsum[8];
#pragma unroll
  for (int r = 0; r < 8; ++r) { rmax[r] = -1e30f; rsum[r] = 0.f; }
#pragma unroll
  for (int dt = 0; dt < 4; ++dt) Oa[dt] = zero;

  for (int j0 = 0; j0 < NT_; j0 += 32) {
    int fc = frame_of(j0);
    if (fc > maxfq) {                 // whole chunk masked
      if (j0 < N1_) { j0 = N1_ - 32; continue; }  // jump to audio keys
      break;                                      // audio frames only grow
    }
    // S = Q * K^T  (two 16-key column tiles, K-dim 64 = 2 steps)
    v8f S[2];
#pragma unroll
    for (int nt = 0; nt < 2; ++nt) {
      const _Float16* kp = k + qk0 + (size_t)(j0 + nt * 16 + l15) * HD_ + g * 8;
      v16h b0 = cat16(*(const h8*)kp, *(const h8*)(kp + 16));
      v16h b1 = cat16(*(const h8*)(kp + 32), *(const h8*)(kp + 48));
      v8f t = wmma16(Aq[0], b0, zero);
      S[nt] = wmma16(Aq[1], b1, t);
    }
    // online softmax (rows split across the two 16-lane halves)
    float alpha[8];
#pragma unroll
    for (int r = 0; r < 8; ++r) {
      float s0 = S[0][r] * 0.125f, s1 = S[1][r] * 0.125f;
      int f0 = frame_of(j0 + l15), f1 = frame_of(j0 + 16 + l15);
      if (f0 > fq[r] || (fq[r] == FRAMES_ - 1 && f0 == 0)) s0 -= 1e9f;
      if (f1 > fq[r] || (fq[r] == FRAMES_ - 1 && f1 == 0)) s1 -= 1e9f;
      float mx = fmaxf(s0, s1);
      mx = fmaxf(mx, __shfl_xor(mx, 1, 32));
      mx = fmaxf(mx, __shfl_xor(mx, 2, 32));
      mx = fmaxf(mx, __shfl_xor(mx, 4, 32));
      mx = fmaxf(mx, __shfl_xor(mx, 8, 32));
      float mn = fmaxf(rmax[r], mx);
      alpha[r] = __expf(rmax[r] - mn);
      rmax[r] = mn;
      float p0 = __expf(s0 - mn), p1 = __expf(s1 - mn);
      float ps = p0 + p1;
      ps += __shfl_xor(ps, 1, 32); ps += __shfl_xor(ps, 2, 32);
      ps += __shfl_xor(ps, 4, 32); ps += __shfl_xor(ps, 8, 32);
      rsum[r] = rsum[r] * alpha[r] + ps;
      P[(r + g * 8) * 40 + l15]      = (_Float16)p0;
      P[(r + g * 8) * 40 + 16 + l15] = (_Float16)p1;
    }
#pragma unroll
    for (int dt = 0; dt < 4; ++dt)
#pragma unroll
      for (int r = 0; r < 8; ++r) Oa[dt][r] *= alpha[r];
    __syncthreads();
    // O += P(16x32) * V(32x64): P re-read in A-layout from LDS
    const _Float16* pb = &P[l15 * 40 + g * 8];
    v16h Pa = cat16(*(const h8*)pb, *(const h8*)(pb + 16));
#pragma unroll
    for (int dt = 0; dt < 4; ++dt) {
      const _Float16* vp = vT + (size_t)bh * HD_ * NT_ +
                           (size_t)(dt * 16 + l15) * NT_ + j0 + g * 8;
      v16h Bv = cat16(*(const h8*)vp, *(const h8*)(vp + 16));
      Oa[dt] = wmma16(Pa, Bv, Oa[dt]);
    }
    __syncthreads();
  }

  const int b = bh >> 4, h = bh & 15;
#pragma unroll
  for (int dt = 0; dt < 4; ++dt)
#pragma unroll
    for (int r = 0; r < 8; ++r) {
      int m = q0 + r + g * 8;
      int ch = h * HD_ + dt * 16 + l15;
      float o = Oa[dt][r] / rsum[r];
      if (m < N1_) o1[((size_t)b * N1_ + m) * D_ + ch] = (_Float16)o;
      else         o2[((size_t)b * N2_ + (m - N1_)) * D_ + ch] = (_Float16)o;
    }
}

// ---------------------------------------------------------------------------
// RMS-norm + RoPE on q/k, transpose-store v. One wave per (b, token, head).
// ---------------------------------------------------------------------------
__global__ __launch_bounds__(256) void qkvproc(
    const float* __restrict__ qkv, const float* __restrict__ qs,
    const float* __restrict__ ks, _Float16* __restrict__ q16,
    _Float16* __restrict__ k16, _Float16* __restrict__ vT,
    int Ntok, int tokbase, int tpf) {
  int wave = blockIdx.x * 8 + (threadIdx.x >> 5);
  int lane = threadIdx.x & 31;
  int h = wave & 15;
  int t2 = wave >> 4;
  int tok = t2 % Ntok;
  int b = t2 / Ntok;
  if (b >= B_) return;

  const float* base = qkv + (size_t)(b * Ntok + tok) * (3 * D_) + h * HD_;
  float pos = (float)(tok / tpf);
  float ang = pos * __expf(-(float)lane * 0.2878231366f);  // ln(1e4)/32
  float sn, cs;
  __sincosf(ang, &sn, &cs);
  size_t rowq = ((size_t)(b * H_ + h) * NT_ + tokbase + tok) * HD_;

#pragma unroll
  for (int sel = 0; sel < 2; ++sel) {  // 0 = q, 1 = k
    const float* p = base + sel * D_;
    const float* sc = sel ? ks : qs;
    float a = p[lane], bb = p[lane + 32];
    float ss = a * a + bb * bb;
#pragma unroll
    for (int m = 1; m < 32; m <<= 1) ss += __shfl_xor(ss, m, 32);
    float inv = rsqrtf(ss * (1.0f / 64.0f) + 1e-6f);
    float na = a * inv * sc[lane];
    float nb = bb * inv * sc[lane + 32];
    float oa = na * cs - nb * sn;
    float ob = nb * cs + na * sn;
    _Float16* dst = sel ? k16 : q16;
    dst[rowq + lane] = (_Float16)oa;
    dst[rowq + lane + 32] = (_Float16)ob;
  }
  const float* pv = base + 2 * D_;
  size_t vb = (size_t)(b * H_ + h) * HD_ * NT_ + tokbase + tok;
  vT[vb + (size_t)lane * NT_] = (_Float16)pv[lane];
  vT[vb + (size_t)(lane + 32) * NT_] = (_Float16)pv[lane + 32];
}

// ---------------------------------------------------------------------------
// adaLN (no-affine LN * (1+scale) + shift) -> f16
// ---------------------------------------------------------------------------
__global__ __launch_bounds__(256) void adaln_k(
    const float* __restrict__ x, const float* __restrict__ adh,
    _Float16* __restrict__ out, int tokPerB) {
  int row = blockIdx.x;
  int b = row / tokPerB;
  const float* xp = x + (size_t)row * D_;
  int tid = threadIdx.x;
  float v[4], s = 0.f, s2 = 0.f;
#pragma unroll
  for (int i = 0; i < 4; ++i) { v[i] = xp[tid + i * 256]; s += v[i]; s2 += v[i] * v[i]; }
#pragma unroll
  for (int m = 1; m < 32; m <<= 1) { s += __shfl_xor(s, m, 32); s2 += __shfl_xor(s2, m, 32); }
  __shared__ float r1[8], r2[8];
  if ((tid & 31) == 0) { r1[tid >> 5] = s; r2[tid >> 5] = s2; }
  __syncthreads();
  float ts = 0.f, ts2 = 0.f;
#pragma unroll
  for (int j = 0; j < 8; ++j) { ts += r1[j]; ts2 += r2[j]; }
  float mean = ts * (1.0f / D_);
  float var = ts2 * (1.0f / D_) - mean * mean;
  float rstd = rsqrtf(var + 1e-5f);
#pragma unroll
  for (int i = 0; i < 4; ++i) {
    int d = tid + i * 256;
    float scl = 1.0f + adh[(size_t)b * 2 * D_ + D_ + d];
    float sh  = adh[(size_t)b * 2 * D_ + d];
    out[(size_t)row * D_ + d] = (_Float16)((v[i] - mean) * rstd * scl + sh);
  }
}

__global__ __launch_bounds__(256) void ln_k(
    const float* __restrict__ x, const float* __restrict__ lw,
    const float* __restrict__ lb, _Float16* __restrict__ out) {
  int row = blockIdx.x;
  const float* xp = x + (size_t)row * D_;
  int tid = threadIdx.x;
  float v[4], s = 0.f, s2 = 0.f;
#pragma unroll
  for (int i = 0; i < 4; ++i) { v[i] = xp[tid + i * 256]; s += v[i]; s2 += v[i] * v[i]; }
#pragma unroll
  for (int m = 1; m < 32; m <<= 1) { s += __shfl_xor(s, m, 32); s2 += __shfl_xor(s2, m, 32); }
  __shared__ float r1[8], r2[8];
  if ((tid & 31) == 0) { r1[tid >> 5] = s; r2[tid >> 5] = s2; }
  __syncthreads();
  float ts = 0.f, ts2 = 0.f;
#pragma unroll
  for (int j = 0; j < 8; ++j) { ts += r1[j]; ts2 += r2[j]; }
  float mean = ts * (1.0f / D_);
  float var = ts2 * (1.0f / D_) - mean * mean;
  float rstd = rsqrtf(var + 1e-5f);
#pragma unroll
  for (int i = 0; i < 4; ++i) {
    int d = tid + i * 256;
    out[(size_t)row * D_ + d] = (_Float16)((v[i] - mean) * rstd * lw[d] + lb[d]);
  }
}

// cond projections: out[b,o] = dot(silu_cond[b,:], w[o,:]) + bias[o]
__global__ __launch_bounds__(256) void cond_gemv(
    const float* __restrict__ sc, const float* __restrict__ w,
    const float* __restrict__ bias, float* __restrict__ out, int O) {
  int o = blockIdx.x * 8 + (threadIdx.x >> 5);
  int lane = threadIdx.x & 31;
  int b = blockIdx.y;
  float sum = 0.f;
  const float* cb = sc + (size_t)b * D_;
  const float* wb = w + (size_t)o * D_;
#pragma unroll 4
  for (int i = 0; i < 32; ++i) sum += cb[lane + i * 32] * wb[lane + i * 32];
#pragma unroll
  for (int m = 1; m < 32; m <<= 1) sum += __shfl_xor(sum, m, 32);
  if (lane == 0) out[(size_t)b * O + o] = sum + bias[o];
}

__global__ void silu_k(const float* __restrict__ c, float* __restrict__ o, int n) {
  int i = blockIdx.x * 256 + threadIdx.x;
  if (i < n) o[i] = c[i] / (1.0f + __expf(-c[i]));
}

// x[i] += a[i] * (g ? g[b,d] : 1)
__global__ void addres(float* __restrict__ x, const float* __restrict__ a,
                       const float* __restrict__ g, int perB, int n) {
  int i = blockIdx.x * 256 + threadIdx.x;
  if (i >= n) return;
  float v = a[i];
  if (g) {
    int b = i / perB;
    int d = i & (D_ - 1);
    v *= g[b * D_ + d];
  }
  x[i] += v;
}

// ---------------------------------------------------------------------------
extern "C" void kernel_launch(void* const* d_in, const int* in_sizes, int n_in,
                              void* d_out, int out_size, void* d_ws, size_t ws_size,
                              hipStream_t stream) {
  (void)in_sizes; (void)n_in; (void)out_size; (void)ws_size;
  const float* x_in = (const float*)d_in[0];
  const float* y_in = (const float*)d_in[1];
  const float* cond = (const float*)d_in[2];
  auto PRM = [&](int layer, int idx) -> const float* {
    return (const float*)d_in[3 + layer * 32 + idx];
  };
  float* xo = (float*)d_out;                       // running x (B,N1,D)
  float* yo = xo + (size_t)B_ * N1_ * D_;          // running y (B,N2,D)

  char* wp = (char*)d_ws;
  auto take = [&](size_t bytes) -> char* {
    char* p = wp;
    wp += (bytes + 255) & ~(size_t)255;
    return p;
  };
  float*     sc   = (float*)take((size_t)B_ * D_ * 4);
  float*     adh  = (float*)take((size_t)B_ * 2 * D_ * 4);
  float*     gh   = (float*)take((size_t)B_ * D_ * 4);
  _Float16*  xm16 = (_Float16*)take((size_t)B_ * N1_ * D_ * 2);
  _Float16*  ym16 = (_Float16*)take((size_t)B_ * N2_ * D_ * 2);
  size_t qkvBytes = (size_t)B_ * N1_ * 3 * D_ * 4;               // 48 MB
  size_t hidBytes = ((size_t)B_ * N1_ * 4 * D_ + (size_t)B_ * N2_ * 4 * D_) * 2;
  char* big = take(qkvBytes > hidBytes ? qkvBytes : hidBytes);
  float*    qkvx = (float*)big;                                  // phase A
  _Float16* hx16 = (_Float16*)big;                               // phase B
  _Float16* hy16 = (_Float16*)(big + (size_t)B_ * N1_ * 4 * D_ * 2);
  float*    qkvy = (float*)take((size_t)B_ * N2_ * 3 * D_ * 4);
  _Float16* q16  = (_Float16*)take((size_t)B_ * H_ * NT_ * HD_ * 2);
  _Float16* k16  = (_Float16*)take((size_t)B_ * H_ * NT_ * HD_ * 2);
  _Float16* vT   = (_Float16*)take((size_t)B_ * H_ * NT_ * HD_ * 2);
  _Float16* o1h  = (_Float16*)take((size_t)B_ * N1_ * D_ * 2);
  _Float16* o2h  = (_Float16*)take((size_t)B_ * N2_ * D_ * 2);
  float*    a1   = (float*)take((size_t)B_ * N1_ * D_ * 4);
  float*    a2   = (float*)take((size_t)B_ * N2_ * D_ * 4);

  hipMemcpyAsync(xo, x_in, (size_t)B_ * N1_ * D_ * 4, hipMemcpyDeviceToDevice, stream);
  hipMemcpyAsync(yo, y_in, (size_t)B_ * N2_ * D_ * 4, hipMemcpyDeviceToDevice, stream);
  silu_k<<<(B_ * D_ + 255) / 256, 256, 0, stream>>>(cond, sc, B_ * D_);

  const int Mx = B_ * N1_;   // 4096
  const int My = B_ * N2_;   // 256

  for (int l = 0; l < 2; ++l) {
    // ---- attention half ----
    cond_gemv<<<dim3(2 * D_ / 8, B_), 256, 0, stream>>>(sc, PRM(l, 12), PRM(l, 13), adh, 2 * D_);
    cond_gemv<<<dim3(D_ / 8, B_), 256, 0, stream>>>(sc, PRM(l, 16), PRM(l, 17), gh, D_);
    adaln_k<<<Mx, 256, 0, stream>>>(xo, adh, xm16, N1_);
    ln_k<<<My, 256, 0, stream>>>(yo, PRM(l, 20), PRM(l, 21), ym16);

    gemm_wmma<0><<<dim3(3 * D_ / 64, Mx / 128), 256, 0, stream>>>(
        xm16, PRM(l, 0), PRM(l, 1), qkvx, nullptr, Mx, 3 * D_, D_);
    gemm_wmma<0><<<dim3(3 * D_ / 64, My / 128), 256, 0, stream>>>(
        ym16, PRM(l, 2), PRM(l, 3), qkvy, nullptr, My, 3 * D_, D_);

    qkvproc<<<B_ * N1_ * H_ * 32 / 256, 256, 0, stream>>>(
        qkvx, PRM(l, 8), PRM(l, 9), q16, k16, vT, N1_, 0, 64);
    qkvproc<<<B_ * N2_ * H_ * 32 / 256, 256, 0, stream>>>(
        qkvy, PRM(l, 10), PRM(l, 11), q16, k16, vT, N2_, N1_, 4);

    flash_attn<<<dim3(B_ * H_, NT_ / 16), 32, 0, stream>>>(q16, k16, vT, o1h, o2h);

    gemm_wmma<0><<<dim3(D_ / 64, Mx / 128), 256, 0, stream>>>(
        o1h, PRM(l, 4), PRM(l, 5), a1, nullptr, Mx, D_, D_);
    gemm_wmma<0><<<dim3(D_ / 64, My / 128), 256, 0, stream>>>(
        o2h, PRM(l, 6), PRM(l, 7), a2, nullptr, My, D_, D_);

    addres<<<(Mx * D_ + 255) / 256, 256, 0, stream>>>(xo, a1, gh, N1_ * D_, Mx * D_);
    addres<<<(My * D_ + 255) / 256, 256, 0, stream>>>(yo, a2, nullptr, N2_ * D_, My * D_);

    // ---- MLP half ----
    cond_gemv<<<dim3(2 * D_ / 8, B_), 256, 0, stream>>>(sc, PRM(l, 14), PRM(l, 15), adh, 2 * D_);
    cond_gemv<<<dim3(D_ / 8, B_), 256, 0, stream>>>(sc, PRM(l, 18), PRM(l, 19), gh, D_);
    adaln_k<<<Mx, 256, 0, stream>>>(xo, adh, xm16, N1_);
    ln_k<<<My, 256, 0, stream>>>(yo, PRM(l, 22), PRM(l, 23), ym16);

    gemm_wmma<1><<<dim3(4 * D_ / 64, Mx / 128), 256, 0, stream>>>(
        xm16, PRM(l, 24), PRM(l, 25), nullptr, hx16, Mx, 4 * D_, D_);
    gemm_wmma<1><<<dim3(4 * D_ / 64, My / 128), 256, 0, stream>>>(
        ym16, PRM(l, 28), PRM(l, 29), nullptr, hy16, My, 4 * D_, D_);

    gemm_wmma<0><<<dim3(D_ / 64, Mx / 128), 256, 0, stream>>>(
        hx16, PRM(l, 26), PRM(l, 27), a1, nullptr, Mx, D_, 4 * D_);
    gemm_wmma<0><<<dim3(D_ / 64, My / 128), 256, 0, stream>>>(
        hy16, PRM(l, 30), PRM(l, 31), a2, nullptr, My, D_, 4 * D_);

    addres<<<(Mx * D_ + 255) / 256, 256, 0, stream>>>(xo, a1, gh, N1_ * D_, Mx * D_);
    addres<<<(My * D_ + 255) / 256, 256, 0, stream>>>(yo, a2, nullptr, N2_ * D_, My * D_);
  }
}
